// MGEGFP_58213986730288
// MI455X (gfx1250) — matrix operand order repaired and checked by reading, not compiled, not run on hardware
//
#include <hip/hip_runtime.h>

// ---------------- problem constants (fixed by the reference) ----------------
#define NVIEW 6
#define NNODE 6400
#define DDIM  2048
#define HDIM  512
#define CDIM  256
#define FDIM  768          // CDIM + HDIM
#define NEDGE 204800

typedef __bf16 bf16_t;
typedef __attribute__((ext_vector_type(16))) __bf16 v16bf;
typedef __attribute__((ext_vector_type(8)))  float  v8f;

// fp32 -> bf16 bits, round-to-nearest-even
__device__ __forceinline__ unsigned bf16r(float f) {
  union { float f; unsigned u; } a; a.f = f;
  unsigned r = a.u + 0x7FFFu + ((a.u >> 16) & 1u);
  return r >> 16;
}
__device__ __forceinline__ bf16_t f2bf(float f) {
  union { unsigned short s; bf16_t b; } o;
  o.s = (unsigned short)bf16r(f);
  return o.b;
}

// ---------------- elementwise helpers ----------------
__global__ void cvt_f32_bf16(const float* __restrict__ in, bf16_t* __restrict__ out, int n) {
  int i = blockIdx.x * blockDim.x + threadIdx.x;
  if (i < n) out[i] = f2bf(in[i]);
}

__global__ void fill_zero_int(int* __restrict__ p, int n) {
  int i = blockIdx.x * blockDim.x + threadIdx.x;
  if (i < n) p[i] = 0;
}

// z_v = (hidden + mid + final)/3, written strided into Z[v, :, CDIM:CDIM+HDIM]
__global__ void zv_mean3(const float* __restrict__ h, const float* __restrict__ m,
                         const float* __restrict__ f, float* __restrict__ zout, int n_total) {
  int idx = blockIdx.x * blockDim.x + threadIdx.x;
  if (idx >= n_total) return;
  int n = idx / HDIM, j = idx - n * HDIM;
  zout[(size_t)n * FDIM + j] = (h[idx] + m[idx] + f[idx]) * (1.0f / 3.0f);
}

// ---------------- B pre-pack: f32 [K,N] row-major -> WMMA fragment-major bf16 ----------------
// Layout: dword[((ntile*Ktiles + ktile)*32 + lane)*8 + i]
//   lane l: c = l&15 (column within 16), g = l>>4 (K half-group)
//   dword i holds bf16 pair (k, n), (k+1, n) with k = ktile*32 + g*16 + 2i, n = ntile*16 + c.
// In the GEMM, a lane's 16-bf16 B fragment is 32 contiguous bytes (2x global_load_b128).
__global__ void pack_b_frag(const float* __restrict__ B, unsigned* __restrict__ out,
                            int N, int K) {
  int idx = blockIdx.x * blockDim.x + threadIdx.x;   // output dword id
  int total = (N * K) >> 1;
  if (idx >= total) return;
  const int Ktiles = K >> 5;
  int i    = idx & 7;
  int lane = (idx >> 3) & 31;
  int tid  = idx >> 8;                 // ntile*Ktiles + ktile
  int kt   = tid % Ktiles;
  int nt   = tid / Ktiles;
  int c = lane & 15, g = lane >> 4;
  int k = (kt << 5) + (g << 4) + 2 * i;
  int n = (nt << 4) + c;
  unsigned lo = bf16r(B[(size_t)k * N + n]);
  unsigned hi = bf16r(B[(size_t)(k + 1) * N + n]);
  out[idx] = lo | (hi << 16);
}

// ---------------- WMMA bf16 GEMM: C[M,N] f32 = A[M,K] bf16 @ Bpacked ----------------
// Block = 256 threads = 8 waves laid out 4(M) x 2(N); wave tile = 64x32 (4x2 16x16 tiles);
// block tile = 256x64. Per K-step(32): 8 v_wmma against 8 A-fragment + 4 B-fragment wide loads.
// (Codegen verified: all loads are clause'd global_load_b128, WMMAs pipelined on s_wait_loadcnt.)
#define WM 4
#define WN 2
__global__ void __launch_bounds__(256)
gemm_bf16_wmma(const bf16_t* __restrict__ A, const unsigned* __restrict__ Bp,
               float* __restrict__ C, int M, int N, int K) {
  const int lane = threadIdx.x & 31;
  const int w  = threadIdx.x >> 5;
  const int wm = w & 3;                 // wave position along M (0..3)
  const int wn = w >> 2;                // wave position along N (0..1)
  const int nBlocksN = N >> 6;          // block tile N = 64
  const int bm = (blockIdx.x / nBlocksN) << 8;   // 256 rows per block
  const int bn = (blockIdx.x % nBlocksN) << 6;
  const int r = lane & 15, g = lane >> 4;
  const int Ktiles = K >> 5;

  const v8f vzero = {0.f, 0.f, 0.f, 0.f, 0.f, 0.f, 0.f, 0.f};
  v8f acc[WM][WN];
#pragma unroll
  for (int t = 0; t < WM; ++t)
#pragma unroll
    for (int j = 0; j < WN; ++j) acc[t][j] = vzero;

  const bf16_t* arow[WM];
#pragma unroll
  for (int t = 0; t < WM; ++t)
    arow[t] = A + (size_t)(bm + wm * 64 + t * 16 + r) * K;

  const unsigned* bbase[WN];
#pragma unroll
  for (int j = 0; j < WN; ++j) {
    int nt = (bn >> 4) + wn * WN + j;
    bbase[j] = Bp + ((size_t)nt * Ktiles) * 256 + lane * 8;
  }

  for (int kt = 0; kt < Ktiles; ++kt) {
    const int k0 = kt << 5;
    union { v16bf v; unsigned u[8]; } a[WM], b[WN];
#pragma unroll
    for (int t = 0; t < WM; ++t)
#pragma unroll
      for (int i = 0; i < 8; ++i) {
        const int kk = 2 * i + (g << 4);
        a[t].u[i] = *reinterpret_cast<const unsigned*>(arow[t] + k0 + kk);
      }
#pragma unroll
    for (int j = 0; j < WN; ++j) {
      const unsigned* p = bbase[j] + (size_t)kt * 256;   // 32B contiguous per lane
#pragma unroll
      for (int i = 0; i < 8; ++i) b[j].u[i] = p[i];
    }
#pragma unroll
    for (int t = 0; t < WM; ++t)
#pragma unroll
      for (int j = 0; j < WN; ++j)
        acc[t][j] = __builtin_amdgcn_wmma_f32_16x16x32_bf16(false, a[t].v, false, b[j].v,
                                                            (short)0, acc[t][j], false, false);
  }

#pragma unroll
  for (int t = 0; t < WM; ++t)
#pragma unroll
    for (int j = 0; j < WN; ++j)
#pragma unroll
      for (int jj = 0; jj < 8; ++jj) {
        const int row = bm + wm * 64 + t * 16 + jj + (g << 3);
        const int col = bn + wn * 32 + j * 16 + r;
        C[(size_t)row * N + col] = acc[t][j][jj];
      }
}

// ---------------- CSR build (once per view, reused by all 6 spmms) ----------------
__global__ void hist_dst(const int* __restrict__ dst, int* __restrict__ deg, int E) {
  int i = blockIdx.x * blockDim.x + threadIdx.x;
  if (i < E) atomicAdd(&deg[dst[i]], 1);
}

// exclusive scan over 6400 degrees, single block of 256 threads (25 elems/thread)
__global__ void __launch_bounds__(256)
exscan_deg(const int* __restrict__ deg, int* __restrict__ rs, int* __restrict__ cur) {
  __shared__ int part[256];
  const int t = threadIdx.x;
  const int CH = NNODE / 256;            // 25
  const int base = t * CH;
  int loc[CH];
  int s = 0;
#pragma unroll
  for (int i = 0; i < CH; ++i) { loc[i] = s; s += deg[base + i]; }
  part[t] = s;
  __syncthreads();
  for (int off = 1; off < 256; off <<= 1) {
    int v = (t >= off) ? part[t - off] : 0;
    __syncthreads();
    if (t >= off) part[t] += v;
    __syncthreads();
  }
  const int pre = (t == 0) ? 0 : part[t - 1];
#pragma unroll
  for (int i = 0; i < CH; ++i) {
    int val = pre + loc[i];
    rs[base + i] = val;
    cur[base + i] = val;
  }
  if (t == 255) rs[NNODE] = part[255];
}

__global__ void csr_scatter(const int* __restrict__ src, const int* __restrict__ dst,
                            const float* __restrict__ val, int* __restrict__ cur,
                            int* __restrict__ ssrc, float* __restrict__ sval, int E) {
  int i = blockIdx.x * blockDim.x + threadIdx.x;
  if (i < E) {
    int p = atomicAdd(&cur[dst[i]], 1);
    ssrc[p] = src[i];
    sval[p] = val[i];
  }
}

// ---------------- gather SpMM + fused epilogue (relu / bf16 / strided f32) ----------------
// One block per destination node: edge descriptors are uniform scalar loads, h[src] rows are
// coalesced vector loads (L2-resident), accumulation in registers, no fp atomics, single store.
__global__ void __launch_bounds__(256)
spmm_gather(const int* __restrict__ rs, const int* __restrict__ ssrc,
            const float* __restrict__ sval, const float* __restrict__ hin,
            int F, int do_relu,
            bf16_t* __restrict__ bfout, float* __restrict__ fout, int ostride) {
  const int n  = blockIdx.x;
  const int j0 = rs[n], j1 = rs[n + 1];
  const int f0 = threadIdx.x;
  float a0 = 0.f, a1 = 0.f;
  for (int j = j0; j < j1; ++j) {
    const int   s = ssrc[j];             // uniform -> scalar load
    const float v = sval[j];             // uniform -> scalar load
    const float* hrow = hin + (size_t)s * F;
    a0 += v * hrow[f0];
    if (F == 512) a1 += v * hrow[f0 + 256];
  }
  if (do_relu) { a0 = fmaxf(a0, 0.f); a1 = fmaxf(a1, 0.f); }
  const size_t ib = (size_t)n * F + f0;
  if (bfout) {
    bfout[ib] = f2bf(a0);
    if (F == 512) bfout[ib + 256] = f2bf(a1);
  }
  if (fout) {
    const size_t ob = (size_t)n * ostride + f0;
    fout[ob] = a0;
    if (F == 512) fout[ob + 256] = a1;
  }
}

// ---------------- gated softmax fusion: one block per node, wave-level reductions ----------------
__global__ void __launch_bounds__(256)
gate_fuse(const float* __restrict__ Z,   // [NVIEW, N, FDIM]
          const float* __restrict__ gw,  // [NVIEW, NVIEW, FDIM]
          const float* __restrict__ gb,  // [NVIEW, NVIEW]
          float* __restrict__ zg, int Nn) {
  const int n = blockIdx.x;
  const int t = threadIdx.x;
  const int lane = t & 31;
  const int wv = t >> 5;                  // 8 waves
  __shared__ float sc[NVIEW * NVIEW];

  // 36 (k,i) dot products, one per wave, no block barriers inside
  for (int p = wv; p < NVIEW * NVIEW; p += 8) {
    const int k = p / NVIEW, i = p - k * NVIEW;
    const float* zi = Z + ((size_t)i * Nn + n) * FDIM;
    const float* ww = gw + ((size_t)p) * FDIM;
    float s = 0.f;
    for (int f = lane; f < FDIM; f += 32) s += zi[f] * ww[f];
#pragma unroll
    for (int off = 16; off; off >>= 1) s += __shfl_down(s, off, 32);
    if (lane == 0) sc[p] = s + gb[p];
    (void)k;
  }
  __syncthreads();
  if (t < NVIEW) {                        // softmax over i for k = t
    float m = -1e30f;
    for (int i = 0; i < NVIEW; ++i) m = fmaxf(m, sc[t * NVIEW + i]);
    float e[NVIEW], s = 0.f;
    for (int i = 0; i < NVIEW; ++i) { e[i] = __expf(sc[t * NVIEW + i] - m); s += e[i]; }
    float inv = 1.0f / s;
    for (int i = 0; i < NVIEW; ++i) sc[t * NVIEW + i] = e[i] * inv;
  }
  __syncthreads();
  for (int f = t; f < FDIM; f += 256) {
    float zi[NVIEW];
    for (int i = 0; i < NVIEW; ++i) zi[i] = Z[((size_t)i * Nn + n) * FDIM + f];
    for (int k = 0; k < NVIEW; ++k) {
      float acc = 0.f;
      for (int i = 0; i < NVIEW; ++i) acc += sc[k * NVIEW + i] * zi[i];
      zg[((size_t)k * Nn + n) * FDIM + f] = acc;
    }
  }
}

// ---------------- host orchestration ----------------
static inline int cdiv(int a, int b) { return (a + b - 1) / b; }

extern "C" void kernel_launch(void* const* d_in, const int* in_sizes, int n_in,
                              void* d_out, int out_size, void* d_ws, size_t ws_size,
                              hipStream_t stream) {
  (void)in_sizes; (void)n_in; (void)out_size; (void)ws_size;
  const float* x       = (const float*)d_in[0];   // [6, N, D]
  const float* adj_val = (const float*)d_in[1];   // [6, E]
  const float* Wb      = (const float*)d_in[2];   // [6, D, H]
  const float* Wo      = (const float*)d_in[3];   // [6, H, H]
  const float* Wf      = (const float*)d_in[4];   // [6, H, H]
  const float* Wc0     = (const float*)d_in[5];   // [D, C]
  const float* Wc1     = (const float*)d_in[6];   // [C, C]
  const float* Wc2     = (const float*)d_in[7];   // [C, C]
  const float* gw      = (const float*)d_in[8];   // [6, 6, F]
  const float* gb      = (const float*)d_in[9];   // [6, 6]
  const int* adj_src   = (const int*)d_in[10];    // [6, E]
  const int* adj_dst   = (const int*)d_in[11];    // [6, E]

  float* zg = (float*)d_out;                            // z_global [6, N, 768]
  float* Z  = zg + (size_t)NVIEW * NNODE * FDIM;        // Z        [6, N, 768]

  // carve workspace (256B aligned slabs, ~95 MB total)
  uintptr_t w = (uintptr_t)d_ws;
  auto take = [&](size_t bytes) -> void* {
    void* p = (void*)w; w += (bytes + 255) & ~(size_t)255; return p;
  };
  bf16_t*   xbf  = (bf16_t*)take((size_t)NNODE * DDIM * 2);
  unsigned* wbp  = (unsigned*)take((size_t)DDIM * HDIM * 2);   // packed fragment bf16
  unsigned* wop  = (unsigned*)take((size_t)HDIM * HDIM * 2);
  unsigned* wfp  = (unsigned*)take((size_t)HDIM * HDIM * 2);
  unsigned* wc0p = (unsigned*)take((size_t)DDIM * CDIM * 2);
  unsigned* wc1p = (unsigned*)take((size_t)CDIM * CDIM * 2);
  unsigned* wc2p = (unsigned*)take((size_t)CDIM * CDIM * 2);
  bf16_t*   hbf  = (bf16_t*)take((size_t)NNODE * HDIM * 2);    // bf16 GEMM input (hidden/mid)
  bf16_t*   cbf  = (bf16_t*)take((size_t)NNODE * CDIM * 2);
  float*    t0   = (float*)take((size_t)NNODE * HDIM * 4);     // GEMM out
  float*    t1   = (float*)take((size_t)NNODE * HDIM * 4);     // 'final' spmm out
  float*    hidden = (float*)take((size_t)NNODE * HDIM * 4);
  float*    midb   = (float*)take((size_t)NNODE * HDIM * 4);
  int*      deg  = (int*)take((size_t)NNODE * 4);              // CSR build
  int*      rs   = (int*)take((size_t)(NNODE + 1) * 4);
  int*      cur  = (int*)take((size_t)NNODE * 4);
  int*      ssrc = (int*)take((size_t)NEDGE * 4);
  float*    sval = (float*)take((size_t)NEDGE * 4);

  const int TPB = 256;
  auto pack = [&](const float* B, unsigned* out, int N, int K) {
    pack_b_frag<<<cdiv((N * K) >> 1, TPB), TPB, 0, stream>>>(B, out, N, K);
  };
  auto gemm = [&](const bf16_t* A, const unsigned* Bp, float* Cp, int M, int N, int K) {
    int blocks = (M >> 8) * (N >> 6);               // 256x64 block tiles (exact for our dims)
    gemm_bf16_wmma<<<blocks, 256, 0, stream>>>(A, Bp, Cp, M, N, K);
  };
  auto spmm = [&](const float* hin, int F, int relu, bf16_t* bfout, float* fout, int ostride) {
    spmm_gather<<<NNODE, 256, 0, stream>>>(rs, ssrc, sval, hin, F, relu, bfout, fout, ostride);
  };

  // shared weights: pack once per launch
  pack(Wc0, wc0p, CDIM, DDIM);
  pack(Wc1, wc1p, CDIM, CDIM);
  pack(Wc2, wc2p, CDIM, CDIM);

  for (int v = 0; v < NVIEW; ++v) {
    cvt_f32_bf16<<<cdiv(NNODE * DDIM, TPB), TPB, 0, stream>>>(x + (size_t)v * NNODE * DDIM,
                                                              xbf, NNODE * DDIM);
    pack(Wb + (size_t)v * DDIM * HDIM, wbp, HDIM, DDIM);
    pack(Wo + (size_t)v * HDIM * HDIM, wop, HDIM, HDIM);
    pack(Wf + (size_t)v * HDIM * HDIM, wfp, HDIM, HDIM);

    // --- CSR build for this view's graph (reused by all 6 spmms) ---
    const int*   src = adj_src + (size_t)v * NEDGE;
    const int*   dst = adj_dst + (size_t)v * NEDGE;
    const float* val = adj_val + (size_t)v * NEDGE;
    fill_zero_int<<<cdiv(NNODE, TPB), TPB, 0, stream>>>(deg, NNODE);
    hist_dst<<<cdiv(NEDGE, TPB), TPB, 0, stream>>>(dst, deg, NEDGE);
    exscan_deg<<<1, 256, 0, stream>>>(deg, rs, cur);
    csr_scatter<<<cdiv(NEDGE, TPB), TPB, 0, stream>>>(src, dst, val, cur, ssrc, sval, NEDGE);

    float* Zv = Z + (size_t)v * NNODE * FDIM;

    // --- view-specific 3-layer GCN ---
    gemm(xbf, wbp, t0, NNODE, HDIM, DDIM);
    spmm(t0, HDIM, 1, hbf, hidden, HDIM);
    gemm(hbf, wop, t0, NNODE, HDIM, HDIM);
    spmm(t0, HDIM, 1, hbf, midb, HDIM);
    gemm(hbf, wfp, t0, NNODE, HDIM, HDIM);
    spmm(t0, HDIM, 0, nullptr, t1, HDIM);                       // final (no relu)
    zv_mean3<<<cdiv(NNODE * HDIM, TPB), TPB, 0, stream>>>(hidden, midb, t1,
                                                          Zv + CDIM, NNODE * HDIM);

    // --- shared common 3-layer GCN ---
    gemm(xbf, wc0p, t0, NNODE, CDIM, DDIM);
    spmm(t0, CDIM, 1, cbf, nullptr, 0);
    gemm(cbf, wc1p, t0, NNODE, CDIM, CDIM);
    spmm(t0, CDIM, 1, cbf, nullptr, 0);
    gemm(cbf, wc2p, t0, NNODE, CDIM, CDIM);
    spmm(t0, CDIM, 1, nullptr, Zv, FDIM);                       // relu'd, strided into Z[:, :C]
  }

  // gated softmax fusion -> z_global
  gate_fuse<<<NNODE, 256, 0, stream>>>(Z, gw, gb, zg, NNODE);
}
